// LabelWiseAttn_57320633532497
// MI455X (gfx1250) — matrix operand order repaired and checked by reading, not compiled, not run on hardware
//
#include <hip/hip_runtime.h>

// Problem constants (from reference): B=8, L=2500, D=512, C=8921
#define B_  8
#define L_  2500
#define D_  512
#define C_  8921
#define LP  2528            // L padded to multiple of 32 (zero-filled)
#define CP  8928            // C padded to multiple of 16 (zero rows)
#define CT  (CP / 16)       // 558 label tiles
#define LTILES (LP / 32)    // 79 sequence chunks

typedef __bf16  bf16_t;
typedef bf16_t  v16bf __attribute__((ext_vector_type(16)));
typedef float   v8f   __attribute__((ext_vector_type(8)));

__device__ __forceinline__ unsigned short f32_to_bf16(float f) {
  unsigned int u = __float_as_uint(f);
  u += 0x7fffu + ((u >> 16) & 1u);          // round-to-nearest-even
  return (unsigned short)(u >> 16);
}

// ---------------------------------------------------------------------------
// Phase 0a: x (f32 [B][L][D]) -> bf16 row-major [B][LP][D] and bf16
// transposed [B][D][LP], zero-padded in L.  One-time ~20MB transform; the
// transposed copy makes GEMM2 B-fragments contiguous (no transpose loads).
// ---------------------------------------------------------------------------
__global__ void convert_x_kernel(const float* __restrict__ x,
                                 unsigned short* __restrict__ xr,
                                 unsigned short* __restrict__ xt) {
  long i = (long)blockIdx.x * 256 + threadIdx.x;     // over B*LP*D
  if (i >= (long)B_ * LP * D_) return;
  int  d = (int)(i % D_);
  long t = i / D_;
  int  l = (int)(t % LP);
  int  b = (int)(t / LP);
  float v = (l < L_) ? x[((long)b * L_ + l) * D_ + d] : 0.0f;
  unsigned short h = f32_to_bf16(v);
  xr[i] = h;
  xt[((long)b * D_ + d) * LP + l] = h;
}

// Phase 0b: U (f32 [C][D]) -> bf16 [CP][D], zero rows for c >= C.
__global__ void convert_u_kernel(const float* __restrict__ U,
                                 unsigned short* __restrict__ Up) {
  long i = (long)blockIdx.x * 256 + threadIdx.x;     // over CP*D
  if (i >= (long)CP * D_) return;
  int d = (int)(i % D_);
  int c = (int)(i / D_);
  Up[i] = (c < C_) ? f32_to_bf16(U[(long)c * D_ + d]) : (unsigned short)0;
}

// ---------------------------------------------------------------------------
// Fused label-wise attention, flash-style online softmax over L.
// Block = 128 threads = 4 waves; block owns 16 labels of one batch.
// Wave w owns D-columns [w*128, w*128+128) of the m accumulator.
// ---------------------------------------------------------------------------
__global__ void __launch_bounds__(128)
lwattn_kernel(const unsigned short* __restrict__ xr,   // [B][LP][D] bf16
              const unsigned short* __restrict__ xt,   // [B][D][LP] bf16
              const unsigned short* __restrict__ Up,   // [CP][D]    bf16
              float* __restrict__ out)                 // [B][C][D]  f32
{
  __shared__ float          lds_part[4][2][8][32];  // per-wave att partials
  __shared__ float          lds_att[16][32];        // reduced logits tile
  __shared__ unsigned short lds_alpha[16][32];      // exp() tile, bf16
  __shared__ float          lds_corr[16];           // per-row rescale
  __shared__ float          lds_M[16];              // running row max
  __shared__ float          lds_S[16];              // running row sum

  const int tid    = threadIdx.x;
  const int wave   = tid >> 5;
  const int lane   = tid & 31;
  const int laneHi = lane >> 4;        // 0: lanes 0-15, 1: lanes 16-31
  const int lane15 = lane & 15;

  const int ct     = blockIdx.x % CT;
  const int b      = blockIdx.x / CT;
  const int c0     = ct * 16;
  const int dslice = wave * 128;

  union FA { uint4 q[2]; v16bf v; };

  // U A-fragments for this wave's D slice: loop-invariant, keep in VGPRs.
  // A k order used here (contiguous, same as B): lanes0-15 get K = 0..15; lanes16-31 get K = 16..31.
  FA ufrag[4];
  {
    const unsigned short* urow = Up + (long)(c0 + lane15) * D_;
    #pragma unroll
    for (int kk = 0; kk < 4; ++kk) {
      int dbase = dslice + kk * 32;
      ufrag[kk].q[0] = *(const uint4*)(urow + dbase + (laneHi ? 16 : 0));
      ufrag[kk].q[1] = *(const uint4*)(urow + dbase + (laneHi ? 24 : 8));
    }
  }

  v8f accm[8] = {};                    // 16 labels x 128 d-cols, f32

  if (tid < 16) { lds_M[tid] = -3.0e38f; lds_S[tid] = 0.0f; }
  __syncthreads();

  for (int it = 0; it < LTILES; ++it) {
    const int l0 = it * 32;

    // Prefetch next chunk's lines into cache (global_prefetch_b8).
    if (it + 1 < LTILES) {
      const unsigned short* nrow =
          xr + ((long)b * LP + (l0 + 32 + lane15)) * D_ + dslice;
      __builtin_prefetch(nrow, 0, 0);
      const unsigned short* ncol =
          xt + ((long)b * D_ + (dslice + lane15)) * LP + l0 + 32;
      __builtin_prefetch(ncol, 0, 0);
    }

    // ---- Stage ALL GEMM1 B-fragments first (batched loads -> partial
    //      loadcnt waits instead of a full drain before every WMMA).
    FA b1[2][4];
    #pragma unroll
    for (int t = 0; t < 2; ++t) {
      // B 32x16 as loaded here (contiguous k): lane = N (=l), halves = 16 consecutive K (=d).
      const unsigned short* xrow =
          xr + ((long)b * LP + (l0 + t * 16 + lane15)) * D_;
      #pragma unroll
      for (int kk = 0; kk < 4; ++kk) {
        int dbase = dslice + kk * 32 + (laneHi ? 16 : 0);
        b1[t][kk].q[0] = *(const uint4*)(xrow + dbase);
        b1[t][kk].q[1] = *(const uint4*)(xrow + dbase + 8);
      }
    }

    // ---- GEMM1: partial att[16c x 32l] over this wave's 128-wide D slice.
    v8f acca[2] = {};
    #pragma unroll
    for (int kk = 0; kk < 4; ++kk) {
      #pragma unroll
      for (int t = 0; t < 2; ++t) {
        acca[t] = __builtin_amdgcn_wmma_f32_16x16x32_bf16(
            false, ufrag[kk].v, false, b1[t][kk].v, (short)0, acca[t],
            false, false);
      }
    }
    #pragma unroll
    for (int t = 0; t < 2; ++t)
      #pragma unroll
      for (int r = 0; r < 8; ++r)
        lds_part[wave][t][r][lane] = acca[t][r];

    // ---- Issue GEMM2 B-fragment loads NOW: they depend only on l0, so
    //      their latency hides under the reduction + softmax + barriers.
    FA b2[8];
    #pragma unroll
    for (int t = 0; t < 8; ++t) {
      const unsigned short* xcol =
          xt + ((long)b * D_ + (dslice + t * 16 + lane15)) * LP
             + l0 + (laneHi ? 16 : 0);
      b2[t].q[0] = *(const uint4*)(xcol);
      b2[t].q[1] = *(const uint4*)(xcol + 8);
    }
    __syncthreads();

    // ---- Reduce the 4 wave partials: 512 elements / 128 threads.
    #pragma unroll
    for (int e = 0; e < 4; ++e) {
      int idx = tid + e * 128;                // 0..511
      int c = idx >> 5, l = idx & 31;
      float s = 0.0f;
      #pragma unroll
      for (int w = 0; w < 4; ++w)
        s += lds_part[w][l >> 4][c & 7][(l & 15) | ((c >> 3) << 4)];
      lds_att[c][l] = (l0 + l < L_) ? s : -3.0e38f;   // mask L padding
    }
    __syncthreads();

    // ---- Online softmax: one thread per label row.
    if (tid < 16) {
      float mOld = lds_M[tid];
      float mNew = mOld;
      #pragma unroll
      for (int l = 0; l < 32; ++l) mNew = fmaxf(mNew, lds_att[tid][l]);
      float corr = __expf(mOld - mNew);
      float s = 0.0f;
      #pragma unroll
      for (int l = 0; l < 32; ++l) {
        float e = __expf(lds_att[tid][l] - mNew);
        lds_alpha[tid][l] = f32_to_bf16(e);
        s += e;
      }
      lds_S[tid]    = lds_S[tid] * corr + s;
      lds_M[tid]    = mNew;
      lds_corr[tid] = corr;
    }
    __syncthreads();

    // ---- Rescale running m accumulators by the row correction factors.
    float cr[8];
    #pragma unroll
    for (int r = 0; r < 8; ++r) cr[r] = lds_corr[(laneHi << 3) + r];
    #pragma unroll
    for (int t = 0; t < 8; ++t)
      #pragma unroll
      for (int r = 0; r < 8; ++r)
        accm[t][r] *= cr[r];

    // ---- GEMM2: m[16c x 128d] += alpha[16c x 32l] * x[32l x 128d].
    //      B fragments are already resident (loaded before the barriers).
    FA afrag;
    afrag.q[0] = *(const uint4*)&lds_alpha[lane15][laneHi ? 16 : 0];
    afrag.q[1] = *(const uint4*)&lds_alpha[lane15][laneHi ? 24 : 8];
    #pragma unroll
    for (int t = 0; t < 8; ++t) {
      accm[t] = __builtin_amdgcn_wmma_f32_16x16x32_bf16(
          false, afrag.v, false, b2[t].v, (short)0, accm[t], false, false);
    }
    __syncthreads();   // protect lds_alpha before next iteration rewrites it
  }

  // ---- Finalize: m = acc / rowsum, store f32 (skip padded labels).
  float sr[8];
  #pragma unroll
  for (int r = 0; r < 8; ++r) sr[r] = 1.0f / lds_S[(laneHi << 3) + r];
  #pragma unroll
  for (int t = 0; t < 8; ++t) {
    int d = dslice + t * 16 + lane15;
    #pragma unroll
    for (int r = 0; r < 8; ++r) {
      int c = c0 + (laneHi << 3) + r;
      if (c < C_)
        out[((long)b * C_ + c) * D_ + d] = accm[t][r] * sr[r];
    }
  }
}

// ---------------------------------------------------------------------------
extern "C" void kernel_launch(void* const* d_in, const int* in_sizes, int n_in,
                              void* d_out, int out_size, void* d_ws, size_t ws_size,
                              hipStream_t stream) {
  const float* x = (const float*)d_in[0];    // [B][L][D]
  const float* U = (const float*)d_in[1];    // [C][D]
  float* out = (float*)d_out;                // [B][C][D]

  // Workspace layout (bf16 = unsigned short): xr | xt | Up  (~50.6 MB total)
  unsigned short* xr = (unsigned short*)d_ws;
  unsigned short* xt = xr + (size_t)B_ * LP * D_;
  unsigned short* Up = xt + (size_t)B_ * LP * D_;

  long nx = (long)B_ * LP * D_;
  convert_x_kernel<<<(unsigned)((nx + 255) / 256), 256, 0, stream>>>(x, xr, xt);
  long nu = (long)CP * D_;
  convert_u_kernel<<<(unsigned)((nu + 255) / 256), 256, 0, stream>>>(U, Up);

  lwattn_kernel<<<B_ * CT, 128, 0, stream>>>(xr, xt, Up, out);
}
